// PerformerPred_4793183502654
// MI455X (gfx1250) — matrix operand order repaired and checked by reading, not compiled, not run on hardware
//
#include <hip/hip_runtime.h>

typedef __bf16 bf16;
typedef bf16 v16bf __attribute__((ext_vector_type(16)));
typedef bf16 v8bf  __attribute__((ext_vector_type(8)));
typedef float v8f  __attribute__((ext_vector_type(8)));

constexpr int Bn = 8, Sn = 2048, Dn = 768, Hn = 12, DHn = 64, Mn = 64, FFn = 3072, Ln = 4;
constexpr int Rn = Bn * Sn;                       // 16384 token rows
constexpr float DNF  = 0.35355339059327373f;      // 64^-0.25
constexpr float KEPS = 0.001f;

// ------------------------------------------------------------------
// elementwise helpers
// ------------------------------------------------------------------
__global__ void copy2_kernel(const float4* __restrict__ x, float4* __restrict__ a,
                             float4* __restrict__ b, int n4) {
  int i = blockIdx.x * blockDim.x + threadIdx.x;
  if (i < n4) { float4 v = x[i]; a[i] = v; b[i] = v; }
}

__global__ void avg_kernel(const float4* __restrict__ a, const float4* __restrict__ b,
                           float4* __restrict__ o, int n4) {
  int i = blockIdx.x * blockDim.x + threadIdx.x;
  if (i < n4) {
    float4 u = a[i], v = b[i];
    o[i] = make_float4(0.5f * (u.x + v.x), 0.5f * (u.y + v.y),
                       0.5f * (u.z + v.z), 0.5f * (u.w + v.w));
  }
}

// W[K][N] (f32, row-major) -> Wt[N][K] (bf16)  : column-major-for-B layout
__global__ void wt_kernel(const float* __restrict__ W, bf16* __restrict__ Wt, int K, int N) {
  size_t i = (size_t)blockIdx.x * blockDim.x + threadIdx.x;
  if (i >= (size_t)K * N) return;
  int k = (int)(i / N), n = (int)(i % N);
  Wt[(size_t)n * K + k] = (bf16)W[i];
}

// ------------------------------------------------------------------
// token-shift (+optional keep-mask, mask = ids==0) + layernorm -> bf16
// one block per token row; 256 threads x 3 features
// ------------------------------------------------------------------
__global__ __launch_bounds__(256)
void lnshift_kernel(const float* __restrict__ xin, const int* __restrict__ ids,
                    const float* __restrict__ g, const float* __restrict__ bl,
                    bf16* __restrict__ hout, int useMask) {
  const int row = blockIdx.x;
  const int b = row / Sn, s = row % Sn;
  const int t = threadIdx.x;
  float vals[3];
#pragma unroll
  for (int c = 0; c < 3; ++c) {
    const int d = c * 256 + t;
    float v;
    if (c == 1) {
      v = xin[(size_t)row * Dn + d];
    } else {
      const int ss = (c == 0) ? (s + 1) : (s - 1);   // SHIFTS = (-1, 0, 1)
      bool ok = (ss >= 0 && ss < Sn);
      if (ok && useMask) ok = (ids[b * Sn + ss] == 0);  // keep where mask True
      v = ok ? xin[((size_t)(b * Sn + ss)) * Dn + d] : 0.f;
    }
    vals[c] = v;
  }
  __shared__ float rs[256], rq[256];
  float su = vals[0] + vals[1] + vals[2];
  float sq = vals[0] * vals[0] + vals[1] * vals[1] + vals[2] * vals[2];
  rs[t] = su; rq[t] = sq;
  __syncthreads();
  for (int st = 128; st > 0; st >>= 1) {
    if (t < st) { rs[t] += rs[t + st]; rq[t] += rq[t + st]; }
    __syncthreads();
  }
  const float mu  = rs[0] * (1.0f / Dn);
  const float var = rq[0] * (1.0f / Dn) - mu * mu;
  const float inv = rsqrtf(var + 1e-5f);
#pragma unroll
  for (int c = 0; c < 3; ++c) {
    const int d = c * 256 + t;
    float h = (vals[c] - mu) * inv * g[d] + bl[d];
    hout[(size_t)row * Dn + d] = (bf16)h;
  }
}

// ------------------------------------------------------------------
// bf16 WMMA GEMM: out[R][N] = A[R][K] * Wt[N][K]^T + bias, epilogues:
//   EPI 0: write f32   EPI 1: += residual, write f32 (in-place safe)
//   EPI 2: exact GELU, write bf16
// Block tile 128x256x32, 8 waves (2x4), wave tile 64x64 = 4x4 WMMA frags
// Double-buffered LDS + register prefetch, one barrier per k-step.
// ------------------------------------------------------------------
__device__ inline v16bf ld_frag(const bf16* p0, const bf16* p1) {
  union { v16bf v; v8bf h[2]; } u;
  u.h[0] = *(const v8bf*)p0;
  u.h[1] = *(const v8bf*)p1;
  return u.v;
}

template <int EPI>
__global__ __launch_bounds__(256, 1)
void gemm_kernel(const bf16* __restrict__ A, const bf16* __restrict__ Wt,
                 const float* __restrict__ bias, const float* __restrict__ res,
                 float* __restrict__ outF, bf16* __restrict__ outB, int K, int N) {
  __shared__ bf16 As[2][128][40];   // 80B row stride: 16B aligned, bank-spread
  __shared__ bf16 Bs[2][256][40];

  const int t = threadIdx.x;
  const int lane = t & 31;
  const int wid = t >> 5;
  const int wm = (wid >> 2) * 64;    // 0 | 64
  const int wn = (wid & 3) * 64;     // 0..192
  const int rowTile = blockIdx.y * 128;
  const int nTile   = blockIdx.x * 256;

  const int lr = t >> 1;             // A staging row 0..127
  const int lc = (t & 1) * 16;       // A staging col 0|16
  const bf16* Aptr = A  + (size_t)(rowTile + lr) * K + lc;
  const bf16* Bptr = Wt + (size_t)(nTile + t) * K;   // B staging: one row/thread

  v8f acc[4][4];
#pragma unroll
  for (int i = 0; i < 4; ++i)
#pragma unroll
    for (int j = 0; j < 4; ++j) acc[i][j] = (v8f){0.f, 0.f, 0.f, 0.f, 0.f, 0.f, 0.f, 0.f};

  const int fr  = lane & 15;                 // fragment row/col within 16
  const int kbA = (lane < 16) ? 0 : 8;       // A-frag K base (ISA 16-bit A layout)
  const int kbB = (lane < 16) ? 0 : 8;       // B-frag K base (same layout as A)

  // ---- prologue: stage tile 0 ----
  {
    union { v16bf v; v8bf h[2]; } ua, ub0, ub1;
    ua.v  = *(const v16bf*)(Aptr);
    ub0.v = *(const v16bf*)(Bptr);
    ub1.v = *(const v16bf*)(Bptr + 16);
    *(v8bf*)&As[0][lr][lc]     = ua.h[0];
    *(v8bf*)&As[0][lr][lc + 8] = ua.h[1];
    *(v8bf*)&Bs[0][t][0]       = ub0.h[0];
    *(v8bf*)&Bs[0][t][8]       = ub0.h[1];
    *(v8bf*)&Bs[0][t][16]      = ub1.h[0];
    *(v8bf*)&Bs[0][t][24]      = ub1.h[1];
  }
  __syncthreads();

  int buf = 0;
  for (int k0 = 0; k0 < K; k0 += 32) {
    const bool haveNext = (k0 + 32 < K);
    union { v16bf v; v8bf h[2]; } ua, ub0, ub1;
    if (haveNext) {     // prefetch next tile into registers
      ua.v  = *(const v16bf*)(Aptr + k0 + 32);
      ub0.v = *(const v16bf*)(Bptr + k0 + 32);
      ub1.v = *(const v16bf*)(Bptr + k0 + 48);
    }

    // ---- compute current buffer: 16 WMMAs ----
    v16bf bf_[4];
#pragma unroll
    for (int j = 0; j < 4; ++j) {
      const bf16* bp = &Bs[buf][wn + j * 16 + fr][kbB];
      bf_[j] = ld_frag(bp, bp + 16);         // K = kbB..+7 , kbB+16..+23
    }
#pragma unroll
    for (int i = 0; i < 4; ++i) {
      const bf16* ap = &As[buf][wm + i * 16 + fr][0];
      v16bf af = ld_frag(ap + kbA, ap + kbA + 16);  // K = kbA..+7 , kbA+16..+23
#pragma unroll
      for (int j = 0; j < 4; ++j)
        acc[i][j] = __builtin_amdgcn_wmma_f32_16x16x32_bf16(
            false, af, false, bf_[j], (short)0, acc[i][j], false, false);
    }

    if (haveNext) {     // stage next tile into the other buffer
      buf ^= 1;
      *(v8bf*)&As[buf][lr][lc]     = ua.h[0];
      *(v8bf*)&As[buf][lr][lc + 8] = ua.h[1];
      *(v8bf*)&Bs[buf][t][0]       = ub0.h[0];
      *(v8bf*)&Bs[buf][t][8]       = ub0.h[1];
      *(v8bf*)&Bs[buf][t][16]      = ub1.h[0];
      *(v8bf*)&Bs[buf][t][24]      = ub1.h[1];
    }
    __syncthreads();
  }

  // epilogue: C frag layout — VGPR p: M=p (lanes<16) / p+8 (lanes>=16), N=lane%16
  const int mOff = (lane < 16) ? 0 : 8;
#pragma unroll
  for (int j = 0; j < 4; ++j) {
    const int col = nTile + wn + j * 16 + fr;
    const float bv = bias[col];
#pragma unroll
    for (int i = 0; i < 4; ++i) {
      const int rbase = rowTile + wm + i * 16 + mOff;
#pragma unroll
      for (int p = 0; p < 8; ++p) {
        const size_t idx = (size_t)(rbase + p) * N + col;
        float v = acc[i][j][p] + bv;
        if (EPI == 0) {
          outF[idx] = v;
        } else if (EPI == 1) {
          outF[idx] = res[idx] + v;
        } else {
          float gg = 0.5f * v * (1.0f + erff(v * 0.70710678118654752f));
          outB[idx] = (bf16)gg;
        }
      }
    }
  }
}

// ------------------------------------------------------------------
// phi: out[b,h,n,m] = relu(dn * q[b,n,h,:] . P[m,:]) + eps
// grid (S/4, H, B), block 256 (m = t&63, token = t>>6)
// ------------------------------------------------------------------
__global__ __launch_bounds__(256)
void phi_kernel(const float* __restrict__ q, const float* __restrict__ P,
                float* __restrict__ outp) {
  __shared__ float Ps[64][65];
  __shared__ float qs[4][64];
  const int t = threadIdx.x;
  const int h = blockIdx.y, b = blockIdx.z;
  { // load P tile
    const int m = t >> 2, cg = (t & 3) * 16;
#pragma unroll
    for (int j = 0; j < 16; ++j) Ps[m][cg + j] = P[m * 64 + cg + j];
  }
  const int tl = t >> 6, m = t & 63;
  const int n = blockIdx.x * 4 + tl;
  qs[tl][m] = q[((size_t)(b * Sn + n)) * Dn + h * 64 + m];
  __syncthreads();
  const float* qr = qs[tl];
  float s = 0.f;
#pragma unroll
  for (int d = 0; d < 64; ++d) s += qr[d] * Ps[m][d];
  float val = fmaxf(s * DNF, 0.f) + KEPS;
  outp[(((size_t)(b * Hn + h)) * Sn + n) * Mn + m] = val;
}

// ks[bh][m] = sum_n kp[bh][n][m] ; one block per (b,h)
__global__ __launch_bounds__(256)
void ks_kernel(const float* __restrict__ kp, float* __restrict__ ks) {
  const int bh = blockIdx.x, t = threadIdx.x;
  const int m = t & 63, c = t >> 6;
  float s = 0.f;
  for (int n = c * 512; n < (c + 1) * 512; ++n)
    s += kp[((size_t)bh * Sn + n) * Mn + m];
  __shared__ float red[256];
  red[t] = s;
  __syncthreads();
  if (c == 0) ks[bh * 64 + m] = red[m] + red[64 + m] + red[128 + m] + red[192 + m];
}

// dinv[i] = 1 / (qp[i,:] . ks[bh,:]),  i = bh*S + n
__global__ __launch_bounds__(256)
void dinv_kernel(const float* __restrict__ qp, const float* __restrict__ ks,
                 float* __restrict__ dinv) {
  const int i = blockIdx.x * 256 + threadIdx.x;   // < 96*2048
  const int bh = i / Sn;
  const float* qr = qp + (size_t)i * 64;
  const float* kr = ks + bh * 64;
  float s = 0.f;
#pragma unroll
  for (int m = 0; m < 64; ++m) s += qr[m] * kr[m];
  dinv[i] = 1.f / s;
}

// ctx[bh][m][d] = sum_n kp[bh][n][m] * (ids[b,n]==0 ? v[b,n,h,d] : 0)
// one block per (b,h); 64-token chunks staged in LDS
__global__ __launch_bounds__(256)
void ctx_kernel(const float* __restrict__ kp, const float* __restrict__ v,
                const int* __restrict__ ids, float* __restrict__ ctx) {
  const int bh = blockIdx.x;
  const int b = bh / Hn, h = bh % Hn;
  __shared__ float kps[64][68];
  __shared__ float vs[64][68];
  const int t = threadIdx.x;
  const int nl = t >> 2, cg = (t & 3) * 16;     // staging: token nl, 16 cols
  const int m = t & 63, dg = (t >> 6) * 16;     // compute: m, d-group
  float acc[16];
#pragma unroll
  for (int k = 0; k < 16; ++k) acc[k] = 0.f;

  for (int n0 = 0; n0 < Sn; n0 += 64) {
    const float* kpp = kp + ((size_t)bh * Sn + n0 + nl) * Mn + cg;
    const float* vp  = v  + ((size_t)(b * Sn + n0 + nl)) * Dn + h * 64 + cg;
    const float keep = (ids[b * Sn + n0 + nl] == 0) ? 1.f : 0.f;
#pragma unroll
    for (int j = 0; j < 16; ++j) { kps[nl][cg + j] = kpp[j]; vs[nl][cg + j] = keep * vp[j]; }
    __syncthreads();
    for (int n = 0; n < 64; ++n) {
      const float kv = kps[n][m];
      const float* vr = &vs[n][dg];
#pragma unroll
      for (int k = 0; k < 16; ++k) acc[k] += kv * vr[k];
    }
    __syncthreads();
  }
  float* cp = ctx + ((size_t)bh * 64 + m) * 64 + dg;
#pragma unroll
  for (int k = 0; k < 16; ++k) cp[k] = acc[k];
}

// ho[b,n,h,d] = dinv[bh,n] * sum_m ctx[bh][m][d] * qp[bh][n][m]  (bf16 out)
// grid (S/64, H, B)
__global__ __launch_bounds__(256)
void out_kernel(const float* __restrict__ qp, const float* __restrict__ ctx,
                const float* __restrict__ dinv, bf16* __restrict__ ho) {
  const int b = blockIdx.z, h = blockIdx.y;
  const int bh = b * Hn + h;
  const int n0 = blockIdx.x * 64;
  __shared__ float cs[64][65];
  const int t = threadIdx.x;
  { const int m = t >> 2, cg = (t & 3) * 16;
    const float* cp = ctx + ((size_t)bh * 64 + m) * 64 + cg;
#pragma unroll
    for (int j = 0; j < 16; ++j) cs[m][cg + j] = cp[j];
  }
  __syncthreads();
  const int d = t & 63, ng = t >> 6;
  for (int k = 0; k < 16; ++k) {
    const int n = n0 + ng * 16 + k;
    const float* qr = qp + ((size_t)bh * Sn + n) * Mn;
    float s = 0.f;
#pragma unroll
    for (int m = 0; m < 64; ++m) s += qr[m] * cs[m][d];
    float val = s * dinv[(size_t)bh * Sn + n];
    ho[((size_t)(b * Sn + n)) * Dn + h * 64 + d] = (bf16)val;
  }
}

// ------------------------------------------------------------------
extern "C" void kernel_launch(void* const* d_in, const int* in_sizes, int n_in,
                              void* d_out, int out_size, void* d_ws, size_t ws_size,
                              hipStream_t stream) {
  const float* x     = (const float*)d_in[0];
  const int*   ids   = (const int*)  d_in[1];
  const float* Wq    = (const float*)d_in[2];
  const float* bq    = (const float*)d_in[3];
  const float* Wk    = (const float*)d_in[4];
  const float* bk    = (const float*)d_in[5];
  const float* Wv    = (const float*)d_in[6];
  const float* bv    = (const float*)d_in[7];
  const float* Wo    = (const float*)d_in[8];
  const float* bo    = (const float*)d_in[9];
  const float* ln1g  = (const float*)d_in[10];
  const float* ln1b  = (const float*)d_in[11];
  const float* W1    = (const float*)d_in[12];
  const float* b1    = (const float*)d_in[13];
  const float* W2    = (const float*)d_in[14];
  const float* b2    = (const float*)d_in[15];
  const float* ln2g  = (const float*)d_in[16];
  const float* ln2b  = (const float*)d_in[17];
  const float* proj  = (const float*)d_in[18];

  char* ws = (char*)d_ws;
  size_t off = 0;
  auto alloc = [&](size_t bytes) -> char* {
    char* p = ws + off;
    off = (off + bytes + 255) & ~(size_t)255;
    return p;
  };
  const size_t RD4 = (size_t)Rn * Dn * 4;
  const size_t RD2 = (size_t)Rn * Dn * 2;
  const size_t BHS = (size_t)Bn * Hn * Sn;

  float* x1    = (float*)alloc(RD4);
  float* x2    = (float*)alloc(RD4);
  bf16*  hb    = (bf16*) alloc(RD2);
  float* qbuf  = (float*)alloc(RD4);
  float* kbuf  = (float*)alloc(RD4);
  float* vbuf  = (float*)alloc(RD4);
  float* qp    = (float*)alloc(BHS * Mn * 4);
  float* kp    = (float*)alloc(BHS * Mn * 4);
  float* ksb   = (float*)alloc((size_t)Bn * Hn * Mn * 4);
  float* dinvb = (float*)alloc(BHS * 4);
  float* ctxb  = (float*)alloc((size_t)Bn * Hn * Mn * DHn * 4);
  bf16*  ho    = (bf16*) alloc(RD2);
  bf16*  ffh   = (bf16*) alloc((size_t)Rn * FFn * 2);
  bf16*  Wqt   = (bf16*) alloc((size_t)Dn * Dn * 2);
  bf16*  Wkt   = (bf16*) alloc((size_t)Dn * Dn * 2);
  bf16*  Wvt   = (bf16*) alloc((size_t)Dn * Dn * 2);
  bf16*  Wot   = (bf16*) alloc((size_t)Dn * Dn * 2);
  bf16*  W1t   = (bf16*) alloc((size_t)Dn * FFn * 2);
  bf16*  W2t   = (bf16*) alloc((size_t)FFn * Dn * 2);

  const int n4 = Rn * Dn / 4;
  copy2_kernel<<<(n4 + 255) / 256, 256, 0, stream>>>((const float4*)x, (float4*)x1,
                                                     (float4*)x2, n4);

  const int gw_dd = (Dn * Dn + 255) / 256;        // 2304
  const int gw_df = (Dn * FFn + 255) / 256;       // 9216
  const dim3 gD(Dn / 256, Rn / 128);              // (3,128)
  const dim3 gF(FFn / 256, Rn / 128);             // (12,128)

  for (int l = 0; l < Ln; ++l) {
    const float* Wq_l = Wq + (size_t)l * Dn * Dn;
    const float* Wk_l = Wk + (size_t)l * Dn * Dn;
    const float* Wv_l = Wv + (size_t)l * Dn * Dn;
    const float* Wo_l = Wo + (size_t)l * Dn * Dn;
    const float* W1_l = W1 + (size_t)l * Dn * FFn;
    const float* W2_l = W2 + (size_t)l * FFn * Dn;
    const float* P_l  = proj + (size_t)l * Mn * DHn;

    wt_kernel<<<gw_dd, 256, 0, stream>>>(Wq_l, Wqt, Dn, Dn);
    wt_kernel<<<gw_dd, 256, 0, stream>>>(Wk_l, Wkt, Dn, Dn);
    wt_kernel<<<gw_dd, 256, 0, stream>>>(Wv_l, Wvt, Dn, Dn);
    wt_kernel<<<gw_dd, 256, 0, stream>>>(Wo_l, Wot, Dn, Dn);
    wt_kernel<<<gw_df, 256, 0, stream>>>(W1_l, W1t, Dn, FFn);
    wt_kernel<<<gw_df, 256, 0, stream>>>(W2_l, W2t, FFn, Dn);

    // ---- attention branch: y1 = x1 + attn(x2) ----
    lnshift_kernel<<<Rn, 256, 0, stream>>>(x2, ids, ln1g + l * Dn, ln1b + l * Dn, hb, 1);
    gemm_kernel<0><<<gD, 256, 0, stream>>>(hb, Wqt, bq + l * Dn, nullptr, qbuf, nullptr, Dn, Dn);
    gemm_kernel<0><<<gD, 256, 0, stream>>>(hb, Wkt, bk + l * Dn, nullptr, kbuf, nullptr, Dn, Dn);
    gemm_kernel<0><<<gD, 256, 0, stream>>>(hb, Wvt, bv + l * Dn, nullptr, vbuf, nullptr, Dn, Dn);

    phi_kernel<<<dim3(Sn / 4, Hn, Bn), 256, 0, stream>>>(qbuf, P_l, qp);
    phi_kernel<<<dim3(Sn / 4, Hn, Bn), 256, 0, stream>>>(kbuf, P_l, kp);
    ks_kernel<<<Bn * Hn, 256, 0, stream>>>(kp, ksb);
    dinv_kernel<<<(int)(BHS / 256), 256, 0, stream>>>(qp, ksb, dinvb);
    ctx_kernel<<<Bn * Hn, 256, 0, stream>>>(kp, vbuf, ids, ctxb);
    out_kernel<<<dim3(Sn / 64, Hn, Bn), 256, 0, stream>>>(qp, ctxb, dinvb, ho);

    gemm_kernel<1><<<gD, 256, 0, stream>>>(ho, Wot, bo + l * Dn, x1, x1, nullptr, Dn, Dn);

    // ---- ff branch: y2 = x2 + ff(y1) ----
    lnshift_kernel<<<Rn, 256, 0, stream>>>(x1, ids, ln2g + l * Dn, ln2b + l * Dn, hb, 0);
    gemm_kernel<2><<<gF, 256, 0, stream>>>(hb, W1t, b1 + l * FFn, nullptr, nullptr, ffh, Dn, FFn);
    gemm_kernel<1><<<gD, 256, 0, stream>>>(ffh, W2t, b2 + l * Dn, x2, x2, nullptr, FFn, Dn);
  }

  avg_kernel<<<(n4 + 255) / 256, 256, 0, stream>>>((const float4*)x1, (const float4*)x2,
                                                   (float4*)d_out, n4);
}
